// CosAttentionsMaxNet_12232066859025
// MI455X (gfx1250) — compile-verified, matrix operand order checked by reading, not run on hardware
//
#include <hip/hip_runtime.h>
#include <hip/hip_bf16.h>
#include <math.h>

// ---------------------------------------------------------------------------
// CosAttentionsMaxNet for MI455X (gfx1250, wave32, WMMA).
// All GEMMs in bf16 WMMA (v_wmma_f32_16x16x32_bf16) with f32 accumulation.
// Every GEMM operand is padded so M%32==0, N%32==0, K%32==0 -> fragment
// loads are unguarded aligned b128 loads (no EXEC-mask guard paths).
// GRU recurrences run in single workgroups (32 waves) with state in LDS.
// Workspace requirement: ~550 MB (bump allocations below).
// ---------------------------------------------------------------------------

typedef __bf16 bf16;
typedef __attribute__((ext_vector_type(16))) __bf16 v16bf;
typedef __attribute__((ext_vector_type(8)))  __bf16 v8bf;
typedef __attribute__((ext_vector_type(8)))  float  v8f;

#define DEVINL __device__ __forceinline__

// ---- WMMA fragment loader (unguarded, aligned) -----------------------------
// 16-bit A/B 16x32 fragment layout (wave32):
//   lane L: row = L&15, half = L>>4
//   elems 0..7  = K[k0 + half*8 .. +7]
//   elems 8..15 = K[k0 + 16 + half*8 .. +7]
// rp = base + row*ld  (per-lane row pointer), requires 16B alignment.
DEVINL v16bf frag_ld(const bf16* rp, int k0, int half) {
  v8bf lo = *(const v8bf*)(rp + k0 + half * 8);
  v8bf hi = *(const v8bf*)(rp + k0 + 16 + half * 8);
  v16bf v;
#pragma unroll
  for (int i = 0; i < 8; ++i) { v[i] = lo[i]; v[i + 8] = hi[i]; }
  return v;
}

DEVINL v8f wmma_bf16(v16bf a, v16bf b, v8f c) {
  return __builtin_amdgcn_wmma_f32_16x16x32_bf16(false, a, false, b,
                                                 (short)0, c, false, false);
}

// ---- Generic batched GEMM: C[M,N] = A[M,K] * Bt[N,K]^T (+bias[N]) ----------
// Requires M%32==0, N%32==0, K%32==0. 32x32 output tile per wave:
// 4 accumulators, 4 fragment loads + 4 WMMAs per 32-deep K step.
__global__ void __launch_bounds__(256)
k_gemm_bf16(const bf16* __restrict__ A, long long sA, int lda,
            const bf16* __restrict__ Bt, long long sB, int ldb,
            float* __restrict__ C, long long sC, int ldc,
            const float* __restrict__ bias, int M, int N, int K) {
  int batch = blockIdx.y;
  A  += (long long)batch * sA;
  Bt += (long long)batch * sB;
  C  += (long long)batch * sC;
  int tm = M >> 5, tn = N >> 5;
  int tile = blockIdx.x * blockDim.y + threadIdx.y;
  if (tile >= tm * tn) return;
  int mt = tile % tm, nt = tile / tm;
  int lane = threadIdx.x;
  int row = lane & 15, half = lane >> 4;

  const bf16* Ap0 = A  + (long long)(mt * 32 + row) * lda;
  const bf16* Ap1 = Ap0 + (long long)16 * lda;
  const bf16* Bp0 = Bt + (long long)(nt * 32 + row) * ldb;
  const bf16* Bp1 = Bp0 + (long long)16 * ldb;

  v8f c00 = {}, c01 = {}, c10 = {}, c11 = {};
#pragma unroll 2
  for (int k0 = 0; k0 < K; k0 += 32) {
    if (k0 + 64 < K) {  // global_prefetch_b8 of upcoming K-chunks
      __builtin_prefetch(Ap0 + k0 + 64, 0, 1);
      __builtin_prefetch(Bp0 + k0 + 64, 0, 1);
    }
    v16bf a0 = frag_ld(Ap0, k0, half);
    v16bf a1 = frag_ld(Ap1, k0, half);
    v16bf b0 = frag_ld(Bp0, k0, half);
    v16bf b1 = frag_ld(Bp1, k0, half);
    c00 = wmma_bf16(a0, b0, c00);
    c01 = wmma_bf16(a0, b1, c01);
    c10 = wmma_bf16(a1, b0, c10);
    c11 = wmma_bf16(a1, b1, c11);
  }

  int n0 = nt * 32 + row, n1 = n0 + 16;
  float bv0 = bias ? bias[n0] : 0.0f;
  float bv1 = bias ? bias[n1] : 0.0f;
#pragma unroll
  for (int r = 0; r < 8; ++r) {
    long long m0 = mt * 32 + r + half * 8;
    long long m1 = m0 + 16;
    C[m0 * ldc + n0] = c00[r] + bv0;
    C[m0 * ldc + n1] = c01[r] + bv1;
    C[m1 * ldc + n0] = c10[r] + bv0;
    C[m1 * ldc + n1] = c11[r] + bv1;
  }
}

// ---- GRU scan: one block per direction, 32 waves, state in LDS -------------
// xp: (64, T, 384) precomputed x@Wih^T + bih. Whh: (384,128) bf16.
// Each wave owns one 16x16 (batch x hidden-col) block of ALL THREE gates, so
// the nonlinearity needs no cross-wave traffic; 12 WMMAs per time step/wave.
// maxout==0: ys (64, T, 256), write cols [dir*128, dir*128+128)
// maxout==1: ys (64, 256) running max over t of h (per direction cols)
__global__ void __launch_bounds__(1024, 1)
k_gru_scan(const float* __restrict__ xp0, const float* __restrict__ xp1,
           const bf16* __restrict__ W0,  const bf16* __restrict__ W1,
           const float* __restrict__ bh0, const float* __restrict__ bh1,
           float* __restrict__ ys, int T, int yld, int maxout) {
  const int dir = blockIdx.x;
  const float* xp  = dir ? xp1 : xp0;
  const bf16*  Whh = dir ? W1  : W0;
  const float* bhh = dir ? bh1 : bh0;
  const int ycol = dir * 128;

  __shared__ float hs[64][128];
  __shared__ bf16  hb[64][136];   // padded row to dodge bank conflicts

  const int lane = threadIdx.x;          // 0..31
  const int wave = threadIdx.y;          // 0..31
  const int row  = lane & 15, half = lane >> 4;
  const int mbase = (wave & 3) * 16;     // batch-row tile
  const int jbase = (wave >> 2) * 16;    // hidden-column tile (16 of 128)
  const int j = jbase + row;

  for (int idx = wave * 32 + lane; idx < 64 * 128; idx += 1024)
    hs[idx >> 7][idx & 127] = 0.0f;
  for (int idx = wave * 32 + lane; idx < 64 * 136; idx += 1024)
    hb[idx / 136][idx % 136] = (bf16)0.0f;
  __syncthreads();

  const bf16* WRp = Whh + (long long)(      jbase + row) * 128;
  const bf16* WZp = Whh + (long long)(128 + jbase + row) * 128;
  const bf16* WNp = Whh + (long long)(256 + jbase + row) * 128;
  const float bR = bhh[      j];
  const float bZ = bhh[128 + j];
  const float bN = bhh[256 + j];
  const bf16* hp = &hb[mbase + row][0];

  float runmax[8];
#pragma unroll
  for (int r = 0; r < 8; ++r) runmax[r] = -3.4e38f;

  for (int s = 0; s < T; ++s) {
    int t = dir ? (T - 1 - s) : s;       // backward dir scans reversed
    v8f aR = {}, aZ = {}, aN = {};
#pragma unroll
    for (int kk = 0; kk < 128; kk += 32) {
      v16bf a  = frag_ld(hp,  kk, half);   // ds_load_b128 x2 from LDS state
      v16bf br = frag_ld(WRp, kk, half);
      v16bf bz = frag_ld(WZp, kk, half);
      v16bf bn = frag_ld(WNp, kk, half);
      aR = wmma_bf16(a, br, aR);
      aZ = wmma_bf16(a, bz, aZ);
      aN = wmma_bf16(a, bn, aN);
    }
    __syncthreads();   // all waves finished reading hb for this step
#pragma unroll
    for (int r = 0; r < 8; ++r) {
      int m = mbase + r + half * 8;
      const float* xrow = xp + ((long long)m * T + t) * 384;
      float xr = xrow[j], xz = xrow[128 + j], xn = xrow[256 + j];
      float rg = 1.0f / (1.0f + __expf(-(xr + aR[r] + bR)));
      float zg = 1.0f / (1.0f + __expf(-(xz + aZ[r] + bZ)));
      float ng = tanhf(xn + rg * (aN[r] + bN));
      float h2 = (1.0f - zg) * ng + zg * hs[m][j];
      hs[m][j] = h2;
      hb[m][j] = (bf16)h2;
      if (maxout) runmax[r] = fmaxf(runmax[r], h2);
      else        ys[((long long)m * T + t) * yld + ycol + j] = h2;
    }
    __syncthreads();   // hb updated before next step's WMMA reads
  }
  if (maxout) {
#pragma unroll
    for (int r = 0; r < 8; ++r) {
      int m = mbase + r + half * 8;
      ys[(long long)m * yld + ycol + j] = runmax[r];
    }
  }
}

// ---- Elementwise / reshape kernels -----------------------------------------
__global__ void k_cast_pad(const float* __restrict__ s, bf16* __restrict__ d,
                           long long rows, int cols, int ldd) {
  long long i = (long long)blockIdx.x * blockDim.x + threadIdx.x;
  if (i >= rows * (long long)ldd) return;
  long long r = i / ldd; int c = (int)(i % ldd);
  d[i] = (c < cols) ? (bf16)s[r * cols + c] : (bf16)0.0f;
}

__global__ void k_unit_cast(const float* __restrict__ s, bf16* __restrict__ d,
                            long long rows, int cols) {
  long long rowi = (long long)blockIdx.x * 8 + (threadIdx.x >> 5);
  int lane = threadIdx.x & 31;
  if (rowi >= rows) return;
  const float* p = s + rowi * (long long)cols;
  float ss = 0.0f;
  for (int c = lane; c < cols; c += 32) { float v = p[c]; ss += v * v; }
  for (int m = 16; m; m >>= 1) ss += __shfl_xor(ss, m);
  float inv = 1.0f / fmaxf(sqrtf(ss), 1e-8f);
  bf16* q = d + rowi * (long long)cols;
  for (int c = lane; c < cols; c += 32) q[c] = (bf16)(p[c] * inv);
}

__global__ void k_transpose_cast(const float* __restrict__ s, bf16* __restrict__ d,
                                 int rows, int cols, long long sS, long long sD) {
  int b = blockIdx.z;
  s += (long long)b * sS; d += (long long)b * sD;
  int i = blockIdx.x * blockDim.x + threadIdx.x;
  if (i >= rows * cols) return;
  int r = i / cols, c = i % cols;
  d[(long long)c * rows + r] = (bf16)s[i];
}

__global__ void k_concat_cast(const float* __restrict__ a, const float* __restrict__ b,
                              bf16* __restrict__ d, long long rows) {
  long long i = (long long)blockIdx.x * blockDim.x + threadIdx.x;
  if (i >= rows * 512) return;
  long long r = i >> 9; int c = (int)(i & 511);
  d[i] = (bf16)((c < 256) ? a[r * 256 + c] : b[r * 256 + (c - 256)]);
}

// softmax over O (axis=1) of att (64,128,512), emit transposed (64,512,128) bf16
__global__ void k_softmax_o_t(const float* __restrict__ att, bf16* __restrict__ out) {
  int i = blockIdx.x * blockDim.x + threadIdx.x;
  if (i >= 64 * 512) return;
  int b = i >> 9, c = i & 511;
  const float* p = att + (long long)b * 128 * 512 + c;
  float mx = -3.4e38f;
  for (int o = 0; o < 128; ++o) mx = fmaxf(mx, p[(long long)o * 512]);
  float sum = 0.0f;
  for (int o = 0; o < 128; ++o) sum += __expf(p[(long long)o * 512] - mx);
  float inv = 1.0f / sum;
  bf16* q = out + ((long long)b * 512 + c) * 128;
  for (int o = 0; o < 128; ++o) q[o] = (bf16)(__expf(p[(long long)o * 512] - mx) * inv);
}

// softmax over C (axis=2): wave per row of att (64*128, 512), emit bf16
__global__ void k_softmax_c(const float* __restrict__ att, bf16* __restrict__ out) {
  int rowi = blockIdx.x * 8 + (threadIdx.x >> 5);
  int lane = threadIdx.x & 31;
  if (rowi >= 64 * 128) return;
  const float* p = att + (long long)rowi * 512;
  float mx = -3.4e38f;
  for (int c = lane; c < 512; c += 32) mx = fmaxf(mx, p[c]);
  for (int m = 16; m; m >>= 1) mx = fmaxf(mx, __shfl_xor(mx, m));
  float sum = 0.0f;
  for (int c = lane; c < 512; c += 32) sum += __expf(p[c] - mx);
  for (int m = 16; m; m >>= 1) sum += __shfl_xor(sum, m);
  float inv = 1.0f / sum;
  bf16* q = out + (long long)rowi * 512;
  for (int c = lane; c < 512; c += 32) q[c] = (bf16)(__expf(p[c] - mx) * inv);
}

__global__ void k_cos(const float* __restrict__ ce, const float* __restrict__ oe,
                      float* __restrict__ logits, int o) {
  int b = blockIdx.x, lane = threadIdx.x;
  const float* cp = ce + b * 256;
  const float* op = oe + b * 256;
  float dot = 0, nc = 0, no = 0;
  for (int c = lane; c < 256; c += 32) {
    float x = cp[c], y = op[c];
    dot += x * y; nc += x * x; no += y * y;
  }
  for (int m = 16; m; m >>= 1) {
    dot += __shfl_xor(dot, m); nc += __shfl_xor(nc, m); no += __shfl_xor(no, m);
  }
  if (lane == 0)
    logits[b * 10 + o] = dot / (fmaxf(sqrtf(nc), 1e-8f) * fmaxf(sqrtf(no), 1e-8f));
}

__global__ void k_softmax_opts(const float* __restrict__ logits, float* __restrict__ out) {
  int b = blockIdx.x * blockDim.x + threadIdx.x;
  if (b >= 64) return;
  const float* p = logits + b * 10;
  float mx = -3.4e38f;
  for (int o = 0; o < 10; ++o) mx = fmaxf(mx, p[o]);
  float e[10], s = 0.0f;
  for (int o = 0; o < 10; ++o) { e[o] = __expf(p[o] - mx); s += e[o]; }
  for (int o = 0; o < 10; ++o) out[b * 10 + o] = e[o] / s;
}

// ---------------------------------------------------------------------------
extern "C" void kernel_launch(void* const* d_in, const int* in_sizes, int n_in,
                              void* d_out, int out_size, void* d_ws, size_t ws_size,
                              hipStream_t stream) {
  (void)in_sizes; (void)n_in; (void)out_size; (void)ws_size;
  const float* context = (const float*)d_in[0];
  const float* options = (const float*)d_in[2];
  const float* rWihf = (const float*)d_in[4];
  const float* rWhhf = (const float*)d_in[5];
  const float* rbihf = (const float*)d_in[6];
  const float* rbhhf = (const float*)d_in[7];
  const float* rWihb = (const float*)d_in[8];
  const float* rWhhb = (const float*)d_in[9];
  const float* rbihb = (const float*)d_in[10];
  const float* rbhhb = (const float*)d_in[11];
  const float* aWihf = (const float*)d_in[12];
  const float* aWhhf = (const float*)d_in[13];
  const float* abihf = (const float*)d_in[14];
  const float* abhhf = (const float*)d_in[15];
  const float* aWihb = (const float*)d_in[16];
  const float* aWhhb = (const float*)d_in[17];
  const float* abihb = (const float*)d_in[18];
  const float* abhhb = (const float*)d_in[19];
  float* out = (float*)d_out;

  constexpr int B = 64, CTX = 512, OPT = 128, NOPT = 10, E = 300, EP = 320, G = 384;
  constexpr long long CR = (long long)B * CTX;   // 32768 context rows
  constexpr long long OR_ = (long long)B * OPT;  // 8192 option rows (per option)

  char* wp = (char*)d_ws;
  auto alloc = [&](size_t bytes) -> void* {
    void* r = (void*)wp; wp += (bytes + 255) & ~(size_t)255; return r;
  };

  // bf16 weights (padded K for Wih)
  bf16* rWihf_b = (bf16*)alloc((size_t)G * EP * 2);
  bf16* rWihb_b = (bf16*)alloc((size_t)G * EP * 2);
  bf16* rWhhf_b = (bf16*)alloc((size_t)G * 128 * 2);
  bf16* rWhhb_b = (bf16*)alloc((size_t)G * 128 * 2);
  bf16* aWihf_b = (bf16*)alloc((size_t)G * 512 * 2);
  bf16* aWihb_b = (bf16*)alloc((size_t)G * 512 * 2);
  bf16* aWhhf_b = (bf16*)alloc((size_t)G * 128 * 2);
  bf16* aWhhb_b = (bf16*)alloc((size_t)G * 128 * 2);
  // activations
  bf16*  ctx_b     = (bf16*) alloc((size_t)CR * EP * 2);
  bf16*  opts_b    = (bf16*) alloc((size_t)B * NOPT * OPT * EP * 2);
  float* ctx_xp_f  = (float*)alloc((size_t)CR * G * 4);
  float* ctx_xp_b  = (float*)alloc((size_t)CR * G * 4);
  float* ctx_outs  = (float*)alloc((size_t)CR * 256 * 4);
  bf16*  ctx_u_b   = (bf16*) alloc((size_t)CR * 256 * 2);
  bf16*  ctx_t_b   = (bf16*) alloc((size_t)B * 256 * CTX * 2);
  float* opt_xp_f  = (float*)alloc((size_t)OR_ * G * 4);
  float* opt_xp_b  = (float*)alloc((size_t)OR_ * G * 4);
  float* opt_outs  = (float*)alloc((size_t)OR_ * 256 * 4);
  bf16*  opt_u_b   = (bf16*) alloc((size_t)OR_ * 256 * 2);
  bf16*  opt_t_b   = (bf16*) alloc((size_t)B * 256 * OPT * 2);
  float* att       = (float*)alloc((size_t)B * OPT * CTX * 4);
  bf16*  att1t_b   = (bf16*) alloc((size_t)B * CTX * OPT * 2);
  bf16*  att2_b    = (bf16*) alloc((size_t)B * OPT * CTX * 2);
  float* attctx    = (float*)alloc((size_t)B * CTX * 256 * 4);
  float* attopt    = (float*)alloc((size_t)B * OPT * 256 * 4);
  bf16*  cc_ctx_b  = (bf16*) alloc((size_t)CR * 512 * 2);
  bf16*  cc_opt_b  = (bf16*) alloc((size_t)OR_ * 512 * 2);
  float* axp_cf    = (float*)alloc((size_t)CR * G * 4);
  float* axp_cb    = (float*)alloc((size_t)CR * G * 4);
  float* axp_of    = (float*)alloc((size_t)OR_ * G * 4);
  float* axp_ob    = (float*)alloc((size_t)OR_ * G * 4);
  float* ctx_enc   = (float*)alloc((size_t)B * 256 * 4);
  float* opt_enc   = (float*)alloc((size_t)B * 256 * 4);
  float* logits    = (float*)alloc((size_t)B * NOPT * 4);

  auto castpad = [&](const float* s, bf16* d, long long rows, int cols, int ldd) {
    long long total = rows * (long long)ldd;
    k_cast_pad<<<(unsigned)((total + 255) / 256), 256, 0, stream>>>(s, d, rows, cols, ldd);
  };
  auto gemm = [&](const bf16* A, long long sA, int lda,
                  const bf16* Bt, long long sB, int ldb,
                  float* C, long long sC, int ldc, const float* bias,
                  int M, int N, int K, int batch) {
    int tiles = (M >> 5) * (N >> 5);
    dim3 grid((tiles + 7) / 8, batch);
    k_gemm_bf16<<<grid, dim3(32, 8), 0, stream>>>(A, sA, lda, Bt, sB, ldb,
                                                  C, sC, ldc, bias, M, N, K);
  };
  auto scan = [&](const float* x0, const float* x1, const bf16* W0, const bf16* W1,
                  const float* b0, const float* b1, float* ys, int T, int maxout) {
    k_gru_scan<<<dim3(2), dim3(32, 32), 0, stream>>>(x0, x1, W0, W1, b0, b1,
                                                     ys, T, 256, maxout);
  };

  // --- weight + input casts ---
  castpad(rWihf, rWihf_b, G, E, EP);
  castpad(rWihb, rWihb_b, G, E, EP);
  castpad(rWhhf, rWhhf_b, G, 128, 128);
  castpad(rWhhb, rWhhb_b, G, 128, 128);
  castpad(aWihf, aWihf_b, G, 512, 512);
  castpad(aWihb, aWihb_b, G, 512, 512);
  castpad(aWhhf, aWhhf_b, G, 128, 128);
  castpad(aWhhb, aWhhb_b, G, 128, 128);
  castpad(context, ctx_b, CR, E, EP);
  castpad(options, opts_b, (long long)B * NOPT * OPT, E, EP);

  // --- context bi-GRU ---
  gemm(ctx_b, 0, EP, rWihf_b, 0, EP, ctx_xp_f, 0, G, rbihf, (int)CR, G, EP, 1);
  gemm(ctx_b, 0, EP, rWihb_b, 0, EP, ctx_xp_b, 0, G, rbihb, (int)CR, G, EP, 1);
  scan(ctx_xp_f, ctx_xp_b, rWhhf_b, rWhhb_b, rbhhf, rbhhb, ctx_outs, CTX, 0);
  k_unit_cast<<<(unsigned)((CR + 7) / 8), 256, 0, stream>>>(ctx_outs, ctx_u_b, CR, 256);
  k_transpose_cast<<<dim3((CTX * 256 + 255) / 256, 1, B), 256, 0, stream>>>(
      ctx_outs, ctx_t_b, CTX, 256, (long long)CTX * 256, (long long)256 * CTX);

  // --- per-option pipeline ---
  for (int o = 0; o < NOPT; ++o) {
    const bf16* optA = opts_b + (long long)o * OPT * EP;
    long long sOpt = (long long)NOPT * OPT * EP;
    gemm(optA, sOpt, EP, rWihf_b, 0, EP, opt_xp_f, (long long)OPT * G, G, rbihf,
         OPT, G, EP, B);
    gemm(optA, sOpt, EP, rWihb_b, 0, EP, opt_xp_b, (long long)OPT * G, G, rbihb,
         OPT, G, EP, B);
    scan(opt_xp_f, opt_xp_b, rWhhf_b, rWhhb_b, rbhhf, rbhhb, opt_outs, OPT, 0);
    k_unit_cast<<<(unsigned)((OR_ + 7) / 8), 256, 0, stream>>>(opt_outs, opt_u_b, OR_, 256);
    k_transpose_cast<<<dim3((OPT * 256 + 255) / 256, 1, B), 256, 0, stream>>>(
        opt_outs, opt_t_b, OPT, 256, (long long)OPT * 256, (long long)256 * OPT);

    // attention scores: att[b] = opt_unit[b] (128,256) . ctx_unit[b]^T (512,256)^T
    gemm(opt_u_b, (long long)OPT * 256, 256, ctx_u_b, (long long)CTX * 256, 256,
         att, (long long)OPT * CTX, CTX, nullptr, OPT, CTX, 256, B);
    k_softmax_o_t<<<(64 * 512 + 255) / 256, 256, 0, stream>>>(att, att1t_b);
    k_softmax_c<<<(64 * 128 + 7) / 8, 256, 0, stream>>>(att, att2_b);

    // att_ctx[b] = att1t[b] (512,128) . opt_outs[b] (128,256)  (Bt = opt_t)
    gemm(att1t_b, (long long)CTX * OPT, OPT, opt_t_b, (long long)256 * OPT, OPT,
         attctx, (long long)CTX * 256, 256, nullptr, CTX, 256, OPT, B);
    // att_opt[b] = att2[b] (128,512) . ctx_outs[b] (512,256)  (Bt = ctx_t)
    gemm(att2_b, (long long)OPT * CTX, CTX, ctx_t_b, (long long)256 * CTX, CTX,
         attopt, (long long)OPT * 256, 256, nullptr, OPT, 256, CTX, B);

    // concat + encoder projections
    k_concat_cast<<<(unsigned)((CR * 512 + 255) / 256), 256, 0, stream>>>(
        attctx, ctx_outs, cc_ctx_b, CR);
    k_concat_cast<<<(unsigned)((OR_ * 512 + 255) / 256), 256, 0, stream>>>(
        attopt, opt_outs, cc_opt_b, OR_);
    gemm(cc_ctx_b, 0, 512, aWihf_b, 0, 512, axp_cf, 0, G, abihf, (int)CR, G, 512, 1);
    gemm(cc_ctx_b, 0, 512, aWihb_b, 0, 512, axp_cb, 0, G, abihb, (int)CR, G, 512, 1);
    gemm(cc_opt_b, 0, 512, aWihf_b, 0, 512, axp_of, 0, G, abihf, (int)OR_, G, 512, 1);
    gemm(cc_opt_b, 0, 512, aWihb_b, 0, 512, axp_ob, 0, G, abihb, (int)OR_, G, 512, 1);

    // encoder bi-GRUs with fused max-over-time
    scan(axp_cf, axp_cb, aWhhf_b, aWhhb_b, abhhf, abhhb, ctx_enc, CTX, 1);
    scan(axp_of, axp_ob, aWhhf_b, aWhhb_b, abhhf, abhhb, opt_enc, OPT, 1);

    k_cos<<<B, 32, 0, stream>>>(ctx_enc, opt_enc, logits, o);
  }

  k_softmax_opts<<<1, 64, 0, stream>>>(logits, out);
}
